// ScaledDotProductAttention_80556406604346
// MI455X (gfx1250) — matrix operand order repaired and checked by reading, not compiled, not run on hardware
//
#include <hip/hip_runtime.h>

typedef __attribute__((ext_vector_type(16))) _Float16 v16h;
typedef __attribute__((ext_vector_type(8)))  _Float16 v8h;
typedef __attribute__((ext_vector_type(4)))  _Float16 v4h;
typedef __attribute__((ext_vector_type(8)))  float    v8f;

#define S_LEN 8192
#define D_DIM 128
#define BM    64            // query rows per block (4 waves x 16)
#define BN    64            // keys per iteration
#define NKB   (S_LEN / BN)  // 128 key blocks
#define NSPLIT 4            // key splits (flash-decoding style)

// ---- 16-lane (half-wave) reductions via DPP16: pure VALU, no LDS traffic ----
template<int CTRL>
static __device__ __forceinline__ float dpp_mov(float x) {
  return __int_as_float(
      __builtin_amdgcn_mov_dpp(__float_as_int(x), CTRL, 0xF, 0xF, true));
}
static __device__ __forceinline__ float red_max16(float x) {
  x = fmaxf(x, dpp_mov<0xB1>(x));    // quad_perm [1,0,3,2]  (xor 1)
  x = fmaxf(x, dpp_mov<0x4E>(x));    // quad_perm [2,3,0,1]  (xor 2)
  x = fmaxf(x, dpp_mov<0x141>(x));   // row_half_mirror      (xor 4 eff.)
  x = fmaxf(x, dpp_mov<0x140>(x));   // row_mirror           (xor 8 eff.)
  return x;
}
static __device__ __forceinline__ float red_sum16(float x) {
  x += dpp_mov<0xB1>(x);
  x += dpp_mov<0x4E>(x);
  x += dpp_mov<0x141>(x);
  x += dpp_mov<0x140>(x);
  return x;
}

// Flash attention, one 16-row query tile per wave32, f16 WMMA with f32 accum.
// Fragment layouts per CDNA5 ISA 7.12.2:
//  A 16x32 : lane L holds M=L&15; half h -> K = (h&7) + 8*(L>=16) + 16*(h>=8)
//  B 32x16 : lane L holds N=L&15; half h -> K = (h&7) + 8*(L>=16) + 16*(h>=8)
//  C 16x16 : vgpr r, lane -> (M = r + 8*(lane>=16), N = lane&15)
template<bool SPLIT>
__launch_bounds__(128)
__global__ void fa_fwd(const float* __restrict__ q, const float* __restrict__ k,
                       const float* __restrict__ v, float* __restrict__ outO,
                       float* __restrict__ wsM, float* __restrict__ wsL,
                       int kbPerSplit) {
  __shared__ __align__(32) _Float16 ldsK[4 * 4 * 32 * 16]; // [u(4)][c(4)][lane][h]
  __shared__ __align__(32) _Float16 ldsV[8 * 2 * 32 * 16]; // [t(8)][g(2)][lane][h]
  __shared__ __align__(32) _Float16 ldsP[4][16 * 64];      // per-wave P, [M][kk]

  const int tid  = threadIdx.x;
  const int wave = tid >> 5;
  const int lane = tid & 31;
  const int hiL  = lane >> 4;
  const int lo16 = lane & 15;

  const int split = blockIdx.y;
  const int kb0   = split * kbPerSplit;
  const int qBase = blockIdx.x * BM + wave * 16;

  const float sc = 0.08838834764831845f; // 1/sqrt(128), folded into Q

  // ---- Q tile -> 4 A fragments (f32*scale -> f16), register resident ----
  v16h aq[4];
  {
    const float* qrow = q + (size_t)(qBase + lo16) * D_DIM;
    #pragma unroll
    for (int c = 0; c < 4; ++c) {
      const int d0 = 32 * c + 8 * hiL;
      const float4 f0 = *(const float4*)(qrow + d0);       // halves 0..3
      const float4 f1 = *(const float4*)(qrow + d0 + 4);   // halves 4..7
      const float4 f2 = *(const float4*)(qrow + d0 + 16);  // halves 8..11
      const float4 f3 = *(const float4*)(qrow + d0 + 20);  // halves 12..15
      aq[c][0]  = (_Float16)(f0.x * sc); aq[c][1]  = (_Float16)(f0.y * sc);
      aq[c][2]  = (_Float16)(f0.z * sc); aq[c][3]  = (_Float16)(f0.w * sc);
      aq[c][4]  = (_Float16)(f1.x * sc); aq[c][5]  = (_Float16)(f1.y * sc);
      aq[c][6]  = (_Float16)(f1.z * sc); aq[c][7]  = (_Float16)(f1.w * sc);
      aq[c][8]  = (_Float16)(f2.x * sc); aq[c][9]  = (_Float16)(f2.y * sc);
      aq[c][10] = (_Float16)(f2.z * sc); aq[c][11] = (_Float16)(f2.w * sc);
      aq[c][12] = (_Float16)(f3.x * sc); aq[c][13] = (_Float16)(f3.y * sc);
      aq[c][14] = (_Float16)(f3.z * sc); aq[c][15] = (_Float16)(f3.w * sc);
    }
  }

  v8f o[8];
  float m_[8], l_[8];
  #pragma unroll
  for (int t = 0; t < 8; ++t) { v8f z = {}; o[t] = z; }
  #pragma unroll
  for (int r = 0; r < 8; ++r) { m_[r] = -3.0e38f; l_[r] = 0.0f; }

  for (int kb = kb0; kb < kb0 + kbPerSplit; ++kb) {
    // ---- stage K: float4 along d -> one ds_store_b64 each ----
    #pragma unroll
    for (int it = 0; it < 16; ++it) {
      const int n  = (tid >> 5) + it * 4;   // key row 0..63
      const int d0 = (tid & 31) * 4;        // 0..124
      const float4 kv = *(const float4*)(k + (size_t)(kb * BN + n) * D_DIM + d0);
      const int u = n >> 4, nl = n & 15;
      const int c = d0 >> 5, dd = d0 & 31;
      const int laneT = nl + 16 * ((dd >> 3) & 1);
      v4h hk;
      hk[0] = (_Float16)kv.x; hk[1] = (_Float16)kv.y;
      hk[2] = (_Float16)kv.z; hk[3] = (_Float16)kv.w;
      *(v4h*)&ldsK[(((u * 4 + c) * 32 + laneT) << 4) + (dd & 7) + 8 * (dd >> 4)] = hk;
    }
    // ---- stage V: 4 rows of one column (coalesced across lanes) -> ds_store_b64 ----
    #pragma unroll
    for (int it = 0; it < 16; ++it) {
      const int n0 = it * 4;          // key rows n0..n0+3, same 16-group
      const int d  = tid;             // 0..127
      const float* vp = v + (size_t)(kb * BN + n0) * D_DIM + d;
      const float e0 = vp[0 * D_DIM], e1 = vp[1 * D_DIM];
      const float e2 = vp[2 * D_DIM], e3 = vp[3 * D_DIM];
      const int t = d >> 4;
      const int g = n0 >> 5;                               // key chunk 0/1
      const int laneT = (d & 15) + 16 * ((n0 >> 3) & 1);
      v4h hv;
      hv[0] = (_Float16)e0; hv[1] = (_Float16)e1;
      hv[2] = (_Float16)e2; hv[3] = (_Float16)e3;
      *(v4h*)&ldsV[(((t * 2 + g) * 32 + laneT) << 4) + (n0 & 7) + 8 * ((n0 >> 4) & 1)] = hv;
    }
    if (kb + 1 < kb0 + kbPerSplit) {  // global_prefetch_b8 for next block
      const size_t pOff =
          (size_t)((kb + 1) * BN + (tid >> 1)) * D_DIM + (tid & 1) * 64;
      __builtin_prefetch(k + pOff, 0, 3);
      __builtin_prefetch(v + pOff, 0, 3);
    }
    __syncthreads();

    // ---- S = Qsc * K^T : 4 column tiles x 4 K-chunks = 16 WMMAs ----
    v8f s[4];
    #pragma unroll
    for (int u = 0; u < 4; ++u) { v8f z = {}; s[u] = z; }
    #pragma unroll
    for (int u = 0; u < 4; ++u) {
      #pragma unroll
      for (int c = 0; c < 4; ++c) {
        const v16h bk = *(const v16h*)&ldsK[((u * 4 + c) * 32 + lane) << 4];
        s[u] = __builtin_amdgcn_wmma_f32_16x16x32_f16(false, aq[c], false, bk,
                                                      (short)0, s[u], false, false);
      }
    }

    // ---- online softmax (row = one 16-lane half; DPP reductions) ----
    float p[4][8], corr[8];
    #pragma unroll
    for (int r = 0; r < 8; ++r) {
      const float mx01 = fmaxf(s[0][r], s[1][r]);
      const float mx23 = fmaxf(s[2][r], s[3][r]);
      const float mx = red_max16(fmaxf(mx01, mx23));
      const float mn = fmaxf(m_[r], mx);
      corr[r] = __expf(m_[r] - mn);
      #pragma unroll
      for (int u = 0; u < 4; ++u) p[u][r] = __expf(s[u][r] - mn);
      const float rs =
          red_sum16((p[0][r] + p[1][r]) + (p[2][r] + p[3][r]));
      l_[r] = l_[r] * corr[r] + rs;
      m_[r] = mn;
    }
    #pragma unroll
    for (int t = 0; t < 8; ++t)
      #pragma unroll
      for (int r = 0; r < 8; ++r)
        o[t][r] *= corr[r];

    // ---- P: C layout -> A layout via per-wave LDS bounce ----
    _Float16* pbuf = &ldsP[wave][0];
    #pragma unroll
    for (int r = 0; r < 8; ++r) {
      const int M = r + 8 * hiL;
      #pragma unroll
      for (int u = 0; u < 4; ++u)
        pbuf[M * 64 + 16 * u + lo16] = (_Float16)p[u][r];
    }
    asm volatile("s_wait_dscnt 0" ::: "memory");  // cross-lane LDS RAW fence

    v16h pa[2];
    #pragma unroll
    for (int g = 0; g < 2; ++g) {
      const v8h plo = *(const v8h*)&pbuf[lo16 * 64 + 32 * g + 8 * hiL];
      const v8h phi = *(const v8h*)&pbuf[lo16 * 64 + 32 * g + 16 + 8 * hiL];
      #pragma unroll
      for (int i = 0; i < 8; ++i) { pa[g][i] = plo[i]; pa[g][8 + i] = phi[i]; }
    }

    // ---- O += P * V : 8 d-tiles x 2 key-chunks = 16 WMMAs ----
    #pragma unroll
    for (int t = 0; t < 8; ++t) {
      #pragma unroll
      for (int g = 0; g < 2; ++g) {
        const v16h bv = *(const v16h*)&ldsV[((t * 2 + g) * 32 + lane) << 4];
        o[t] = __builtin_amdgcn_wmma_f32_16x16x32_f16(false, pa[g], false, bv,
                                                      (short)0, o[t], false, false);
      }
    }

    __syncthreads();
  }

  // ---- epilogue ----
  if (SPLIT) {
    const size_t rowBase = (size_t)split * S_LEN + qBase;
    #pragma unroll
    for (int t = 0; t < 8; ++t)
      #pragma unroll
      for (int r = 0; r < 8; ++r)
        outO[(rowBase + r + 8 * hiL) * D_DIM + 16 * t + lo16] = o[t][r];
    if (lo16 == 0) {
      #pragma unroll
      for (int r = 0; r < 8; ++r) {
        wsM[rowBase + r + 8 * hiL] = m_[r];
        wsL[rowBase + r + 8 * hiL] = l_[r];
      }
    }
  } else {
    float inv[8];
    #pragma unroll
    for (int r = 0; r < 8; ++r) inv[r] = 1.0f / l_[r];
    #pragma unroll
    for (int t = 0; t < 8; ++t)
      #pragma unroll
      for (int r = 0; r < 8; ++r)
        outO[(size_t)(qBase + r + 8 * hiL) * D_DIM + 16 * t + lo16] =
            o[t][r] * inv[r];
  }
}

// Merge NSPLIT partial (O, m, l) triples: out = sum_i w_i O_i / sum_i w_i l_i
__launch_bounds__(256)
__global__ void fa_combine(const float* __restrict__ wsO, const float* __restrict__ wsM,
                           const float* __restrict__ wsL, float* __restrict__ out) {
  const int idx = blockIdx.x * 256 + threadIdx.x;  // 0 .. S*D-1
  const int row = idx >> 7;
  float m[NSPLIT], w[NSPLIT];
  float M = -3.0e38f;
  #pragma unroll
  for (int i = 0; i < NSPLIT; ++i) {
    m[i] = wsM[(size_t)i * S_LEN + row];
    M = fmaxf(M, m[i]);
  }
  float den = 0.0f, num = 0.0f;
  #pragma unroll
  for (int i = 0; i < NSPLIT; ++i) {
    w[i] = __expf(m[i] - M);
    den += w[i] * wsL[(size_t)i * S_LEN + row];
    num += w[i] * wsO[(size_t)i * S_LEN * D_DIM + idx];
  }
  out[idx] = num / den;
}

extern "C" void kernel_launch(void* const* d_in, const int* in_sizes, int n_in,
                              void* d_out, int out_size, void* d_ws, size_t ws_size,
                              hipStream_t stream) {
  (void)in_sizes; (void)n_in; (void)out_size;
  const float* q = (const float*)d_in[0];
  const float* k = (const float*)d_in[1];
  const float* v = (const float*)d_in[2];
  float* out = (float*)d_out;

  const size_t wsNeeded =
      (size_t)NSPLIT * S_LEN * (D_DIM + 2) * sizeof(float);  // ~16.25 MB

  if (d_ws != nullptr && ws_size >= wsNeeded) {
    float* wsO = (float*)d_ws;
    float* wsM = wsO + (size_t)NSPLIT * S_LEN * D_DIM;
    float* wsL = wsM + (size_t)NSPLIT * S_LEN;
    dim3 grid(S_LEN / BM, NSPLIT);     // 128 x 4 blocks, 2048 waves
    fa_fwd<true><<<grid, dim3(128), 0, stream>>>(q, k, v, wsO, wsM, wsL,
                                                 NKB / NSPLIT);
    fa_combine<<<dim3(S_LEN * D_DIM / 256), dim3(256), 0, stream>>>(wsO, wsM,
                                                                    wsL, out);
  } else {
    dim3 grid(S_LEN / BM, 1);
    fa_fwd<false><<<grid, dim3(128), 0, stream>>>(q, k, v, out, nullptr,
                                                  nullptr, NKB);
  }
}
